// QuantPlainRepConv_80290118631435
// MI455X (gfx1250) — 8-bit matrix operand order repaired and checked by reading, compiled, not run on hardware
//
#include <hip/hip_runtime.h>
#include <stdint.h>

typedef __attribute__((ext_vector_type(8))) int v8i;

#define BATCH 8
#define HH 256
#define WW 256
#define CH 32

union AV { v8i v; unsigned long long q[4]; };
union BV { v8i v; int4 h[2]; };

#ifndef __has_builtin
#define __has_builtin(x) 0
#endif
#if __has_builtin(__builtin_amdgcn_global_load_async_to_lds_b32) && \
    __has_builtin(__builtin_amdgcn_s_wait_asynccnt)
#define HAVE_ASYNC 1
#else
#define HAVE_ASYNC 0
#endif

#define GPTR(p) ((__attribute__((address_space(1))) int*)(void*)(p))
#define LPTR(p) ((__attribute__((address_space(3))) int*)(void*)(p))


__global__ void pack_head_k(const float* __restrict__ w, const float* __restrict__ swp,
                            float* __restrict__ out) {
    int i = blockIdx.x * 256 + threadIdx.x;
    if (i >= 32 * 3 * 9) return;
    float sw = swp[0];
    float v = w[i] / sw;
    v = fminf(fmaxf(rintf(v), -127.f), 127.f);
    out[i] = sw * v;
}

__global__ void pack_bb_k(const float* __restrict__ w, const float* __restrict__ swl,
                          int8_t* __restrict__ out) {
    int i = blockIdx.x * 256 + threadIdx.x;
    if (i >= 12 * 9 * 32 * 32) return;
    int l  = i / 9216;
    int r  = i % 9216;
    int t  = r / 1024;
    int r2 = r % 1024;
    int o  = r2 >> 5;
    int ic = r2 & 31;
    float sw = swl[l];
    float v = w[((l * 32 + o) * 32 + ic) * 9 + t] / sw;
    v = fminf(fmaxf(rintf(v), -127.f), 127.f);
    out[i] = (int8_t)(int)v;
}

__global__ void pack_tail_k(const float* __restrict__ w, const float* __restrict__ swp,
                            float* __restrict__ out) {
    int i = blockIdx.x * 256 + threadIdx.x;
    if (i >= 12 * 9 * 32) return;
    int o  = i / 288;
    int r  = i % 288;
    int t  = r >> 5;
    int ic = r & 31;
    float sw = swp[0];
    float v = w[(o * 32 + ic) * 9 + t] / sw;
    v = fminf(fmaxf(rintf(v), -127.f), 127.f);
    out[i] = sw * v;
}


__global__ __launch_bounds__(256) void head_conv_k(
    const float* __restrict__ x, const float* __restrict__ wq,
    const float* __restrict__ bias, const float* __restrict__ act_s_p,
    uint8_t* __restrict__ out) {
    __shared__ float sw_[864];
    __shared__ float sb_[32];
    int tid = threadIdx.x;
    for (int i = tid; i < 864; i += 256) sw_[i] = wq[i];
    if (tid < 32) sb_[tid] = bias[tid];
    __syncthreads();
    float s = act_s_p[0];
    int gid = blockIdx.x * 256 + tid;
    int xc = gid & 255, yr = (gid >> 8) & 255, b = gid >> 16;

    float xv[3][9];
    #pragma unroll
    for (int c = 0; c < 3; c++)
        #pragma unroll
        for (int t = 0; t < 9; t++) {
            int gy = yr + t / 3 - 1, gx = xc + t % 3 - 1;
            float v = 0.f;
            if (gy >= 0 && gy < HH && gx >= 0 && gx < WW) {
                float raw = x[((b * 3 + c) * HH + gy) * WW + gx];
                float q = fminf(fmaxf(rintf(raw / s), 0.f), 255.f);
                v = s * q;
            }
            xv[c][t] = v;
        }
    uint8_t* op = out + ((size_t)(b * HH + yr) * WW + xc) * CH;
    #pragma unroll
    for (int oc = 0; oc < 32; oc++) {
        float acc = sb_[oc];
        const float* wr = &sw_[oc * 27];
        #pragma unroll
        for (int c = 0; c < 3; c++)
            #pragma unroll
            for (int t = 0; t < 9; t++) acc += wr[c * 9 + t] * xv[c][t];
        acc = fmaxf(acc, 0.f);
        float q = fminf(fmaxf(rintf(acc / s), 0.f), 255.f);
        op[oc] = (uint8_t)q;
    }
}


__global__ __launch_bounds__(256) void bb_conv_k(
    const uint8_t* __restrict__ in, uint8_t* __restrict__ outp,
    const int8_t* __restrict__ wq_all, const float* __restrict__ swl, int layer,
    const float* __restrict__ bias, const float* __restrict__ act_s_p) {
    __shared__ __align__(16) uint8_t s_act[18 * 18 * 32];
    __shared__ __align__(16) int8_t  s_w[9 * 32 * 32];

    int tid = threadIdx.x;
    int blk = blockIdx.x;
    int xt = blk & 15, yg = (blk >> 4) & 15, b = blk >> 8;
    int x0 = xt * 16, y0 = yg * 16;

    const uint8_t* wg = (const uint8_t*)wq_all + (size_t)layer * 9216;

#if HAVE_ASYNC
    for (int i = tid; i < 2304; i += 256)
        __builtin_amdgcn_global_load_async_to_lds_b32(
            GPTR(wg + i * 4), LPTR((uint8_t*)s_w + i * 4), 0, 0);
    for (int i = tid; i < 2592; i += 256) {
        int d = i & 7, slot = i >> 3;
        int col = slot % 18, row = slot / 18;
        int gy = y0 - 1 + row, gx = x0 - 1 + col;
        if (gy >= 0 && gy < HH && gx >= 0 && gx < WW) {
            const uint8_t* gp = in + (((size_t)(b * HH + gy) * WW + gx) * CH + d * 4);
            __builtin_amdgcn_global_load_async_to_lds_b32(
                GPTR(gp), LPTR(s_act + i * 4), 0, 0);
        } else {
            ((uint32_t*)s_act)[i] = 0u;
        }
    }
    __builtin_amdgcn_s_wait_asynccnt(0);
#else
    const int* wq32 = (const int*)wg;
    int* sw32 = (int*)s_w;
    for (int i = tid; i < 2304; i += 256) sw32[i] = wq32[i];
    const uint32_t* in32 = (const uint32_t*)in;
    uint32_t* sa32 = (uint32_t*)s_act;
    for (int i = tid; i < 2592; i += 256) {
        int d = i & 7, slot = i >> 3;
        int col = slot % 18, row = slot / 18;
        int gy = y0 - 1 + row, gx = x0 - 1 + col;
        uint32_t v = 0u;
        if (gy >= 0 && gy < HH && gx >= 0 && gx < WW)
            v = in32[((size_t)(b * HH + gy) * WW + gx) * 8 + d];
        sa32[i] = v;
    }
#endif
    __syncthreads();

    int lane = tid & 31, wave = tid >> 5;
    int m = lane & 15, hi = lane >> 4, n = lane & 15;
    int r0 = wave * 2, r1 = wave * 2 + 1;

    v8i acc00 = {0, 0, 0, 0, 0, 0, 0, 0};
    v8i acc01 = {0, 0, 0, 0, 0, 0, 0, 0};
    v8i acc10 = {0, 0, 0, 0, 0, 0, 0, 0};
    v8i acc11 = {0, 0, 0, 0, 0, 0, 0, 0};

    #pragma unroll
    for (int c = 0; c < 5; c++) {
        const int t0 = 2 * c, t1 = 2 * c + 1;
        const int dy0 = t0 / 3 - 1, dx0 = t0 % 3 - 1;

        BV b0u, b1u;
        {
            int wb00 = (t0 * 32 + n) * 32 + hi * 16;
            int wb01 = (t0 * 32 + 16 + n) * 32 + hi * 16;
            b0u.h[0] = *(const int4*)(s_w + wb00);
            b1u.h[0] = *(const int4*)(s_w + wb01);
            if (t1 < 9) {
                int wb10 = (t1 * 32 + n) * 32 + hi * 16;
                int wb11 = (t1 * 32 + 16 + n) * 32 + hi * 16;
                b0u.h[1] = *(const int4*)(s_w + wb10);
                b1u.h[1] = *(const int4*)(s_w + wb11);
            } else {
                b0u.h[1] = make_int4(0, 0, 0, 0);
                b1u.h[1] = make_int4(0, 0, 0, 0);
            }
        }

        AV a0, a1;
        int base00 = ((r0 + 1 + dy0) * 18 + (m + 1 + dx0)) * 32;
        int base10 = ((r1 + 1 + dy0) * 18 + (m + 1 + dx0)) * 32;
        a0.q[0] = *(const unsigned long long*)(s_act + base00 + hi * 16);
        a0.q[1] = *(const unsigned long long*)(s_act + base00 + hi * 16 + 8);
        a1.q[0] = *(const unsigned long long*)(s_act + base10 + hi * 16);
        a1.q[1] = *(const unsigned long long*)(s_act + base10 + hi * 16 + 8);
        if (t1 < 9) {
            const int dy1 = t1 / 3 - 1, dx1 = t1 % 3 - 1;
            int base01 = ((r0 + 1 + dy1) * 18 + (m + 1 + dx1)) * 32;
            int base11 = ((r1 + 1 + dy1) * 18 + (m + 1 + dx1)) * 32;
            a0.q[2] = *(const unsigned long long*)(s_act + base01 + hi * 16);
            a0.q[3] = *(const unsigned long long*)(s_act + base01 + hi * 16 + 8);
            a1.q[2] = *(const unsigned long long*)(s_act + base11 + hi * 16);
            a1.q[3] = *(const unsigned long long*)(s_act + base11 + hi * 16 + 8);
        } else {
            a0.q[2] = 0ull; a0.q[3] = 0ull;
            a1.q[2] = 0ull; a1.q[3] = 0ull;
        }

        acc00 = __builtin_amdgcn_wmma_i32_16x16x64_iu8(false, a0.v, true, b0u.v, acc00, false, false);
        acc01 = __builtin_amdgcn_wmma_i32_16x16x64_iu8(false, a0.v, true, b1u.v, acc01, false, false);
        acc10 = __builtin_amdgcn_wmma_i32_16x16x64_iu8(false, a1.v, true, b0u.v, acc10, false, false);
        acc11 = __builtin_amdgcn_wmma_i32_16x16x64_iu8(false, a1.v, true, b1u.v, acc11, false, false);
    }

    float s = act_s_p[0];
    float sw = swl[layer];
    float sinv = 1.f / s;
    float bv0 = bias[n] * sinv;
    float bv1 = bias[16 + n] * sinv;
    uint8_t* orow0 = outp + ((size_t)(b * HH + y0 + r0) * WW + x0) * CH;
    uint8_t* orow1 = outp + ((size_t)(b * HH + y0 + r1) * WW + x0) * CH;
    #pragma unroll
    for (int r = 0; r < 8; r++) {
        int M = r + hi * 8;
        float v00 = fminf(fmaxf(rintf(fmaxf((float)acc00[r] * sw + bv0, 0.f)), 0.f), 255.f);
        float v01 = fminf(fmaxf(rintf(fmaxf((float)acc01[r] * sw + bv1, 0.f)), 0.f), 255.f);
        float v10 = fminf(fmaxf(rintf(fmaxf((float)acc10[r] * sw + bv0, 0.f)), 0.f), 255.f);
        float v11 = fminf(fmaxf(rintf(fmaxf((float)acc11[r] * sw + bv1, 0.f)), 0.f), 255.f);
        orow0[M * 32 + n]      = (uint8_t)v00;
        orow0[M * 32 + 16 + n] = (uint8_t)v01;
        orow1[M * 32 + n]      = (uint8_t)v10;
        orow1[M * 32 + 16 + n] = (uint8_t)v11;
    }
}


__global__ __launch_bounds__(256) void tail_conv_k(
    const uint8_t* __restrict__ yN, const uint8_t* __restrict__ y0b,
    const float* __restrict__ wtail, const float* __restrict__ tb,
    const float* __restrict__ act_s_p, float* __restrict__ out) {
    __shared__ float swt[12 * 9 * 32];
    __shared__ float sb[12];
    int tid = threadIdx.x;
    for (int i = tid; i < 3456; i += 256) swt[i] = wtail[i];
    if (tid < 12) sb[tid] = tb[tid];
    __syncthreads();
    float s = act_s_p[0];
    int gid = blockIdx.x * 256 + tid;
    int xc = gid & 255, yr = (gid >> 8) & 255, b = gid >> 16;

    float acc[12];
    #pragma unroll
    for (int oc = 0; oc < 12; oc++) acc[oc] = sb[oc];

    const uint32_t* a32 = (const uint32_t*)yN;
    const uint32_t* b32 = (const uint32_t*)y0b;

    #pragma unroll
    for (int t = 0; t < 9; t++) {
        int gy = yr + t / 3 - 1, gx = xc + t % 3 - 1;
        float xv[32];
        if (gy >= 0 && gy < HH && gx >= 0 && gx < WW) {
            size_t base = ((size_t)(b * HH + gy) * WW + gx) * 8;
            #pragma unroll
            for (int d = 0; d < 8; d++) {
                uint32_t va = a32[base + d], vb = b32[base + d];
                #pragma unroll
                for (int k = 0; k < 4; k++) {
                    int qsum = (int)((va >> (8 * k)) & 255u) + (int)((vb >> (8 * k)) & 255u);
                    qsum = qsum > 255 ? 255 : qsum;
                    xv[d * 4 + k] = s * (float)qsum;
                }
            }
        } else {
            #pragma unroll
            for (int ci = 0; ci < 32; ci++) xv[ci] = 0.f;
        }
        #pragma unroll
        for (int oc = 0; oc < 12; oc++) {
            const float* wr = &swt[(oc * 9 + t) * 32];
            float a = 0.f;
            #pragma unroll
            for (int ci = 0; ci < 32; ci++) a += wr[ci] * xv[ci];
            acc[oc] += a;
        }
    }

    #pragma unroll
    for (int oc = 0; oc < 12; oc++) {
        float yv = fmaxf(acc[oc], 0.f);
        float q = fminf(fmaxf(rintf(yv / s), 0.f), 255.f);
        float o = fminf(fmaxf(s * q, 0.f), 255.f);
        int cc = oc >> 2, r1 = (oc >> 1) & 1, r2 = oc & 1;
        out[((size_t)(b * 3 + cc) * 512 + (yr * 2 + r1)) * 512 + (xc * 2 + r2)] = o;
    }
}


extern "C" void kernel_launch(void* const* d_in, const int* in_sizes, int n_in,
                              void* d_out, int out_size, void* d_ws, size_t ws_size,
                              hipStream_t stream) {
    const float* x       = (const float*)d_in[0];
    const float* head_w  = (const float*)d_in[1];
    const float* head_b  = (const float*)d_in[2];
    const float* head_sw = (const float*)d_in[3];
    const float* bb_w    = (const float*)d_in[4];
    const float* bb_b    = (const float*)d_in[5];
    const float* bb_sw   = (const float*)d_in[6];
    const float* tr_w    = (const float*)d_in[7];
    const float* tr_b    = (const float*)d_in[8];
    const float* tr_sw   = (const float*)d_in[9];
    const float* act_s   = (const float*)d_in[10];

    const size_t actBytes = (size_t)BATCH * HH * WW * CH;
    uint8_t* act0 = (uint8_t*)d_ws;
    uint8_t* act1 = act0 + actBytes;
    uint8_t* act2 = act1 + actBytes;
    float*   whead = (float*)(act2 + actBytes);
    int8_t*  wbb   = (int8_t*)(whead + 864);
    float*   wtail = (float*)(wbb + 110592);

    pack_head_k<<<4, 256, 0, stream>>>(head_w, head_sw, whead);
    pack_bb_k<<<(12 * 9216 + 255) / 256, 256, 0, stream>>>(bb_w, bb_sw, wbb);
    pack_tail_k<<<(3456 + 255) / 256, 256, 0, stream>>>(tr_w, tr_sw, wtail);

    head_conv_k<<<(BATCH * HH * WW) / 256, 256, 0, stream>>>(x, whead, head_b, act_s, act0);

    uint8_t* bufs[2] = {act1, act2};
    const uint8_t* cur = act0;
    for (int l = 0; l < 12; l++) {
        uint8_t* o = bufs[l & 1];
        bb_conv_k<<<16 * 16 * BATCH, 256, 0, stream>>>(cur, o, wbb, bb_sw, l,
                                                       bb_b + l * 32, act_s);
        cur = o;
    }
    tail_conv_k<<<(BATCH * HH * WW) / 256, 256, 0, stream>>>(cur, act0, wtail, tr_b,
                                                             act_s, (float*)d_out);
}
